// SelfAttention_7842610282755
// MI455X (gfx1250) — matrix operand order repaired and checked by reading, not compiled, not run on hardware
//
#include <hip/hip_runtime.h>
#include <hip/hip_bf16.h>

#define Bsz  8
#define Sseq 1024
#define Dm   1024
#define Hh   16
#define HDim 64

typedef __bf16 bf16;
typedef bf16  v16bf __attribute__((ext_vector_type(16)));
typedef bf16  v8bf  __attribute__((ext_vector_type(8)));
typedef bf16  v4bf  __attribute__((ext_vector_type(4)));
typedef float v8f   __attribute__((ext_vector_type(8)));

union Frag16 { v16bf v; v8bf h[2]; };

__device__ inline v8f zero8() {
    v8f z;
    #pragma unroll
    for (int i = 0; i < 8; ++i) z[i] = 0.0f;
    return z;
}

// A-fragment (16x32 bf16, M x K): lane holds row m = lane&15;
// K contiguous 16g..16g+15, g = lane>>4.
__device__ inline v16bf load_frag_a(const bf16* row, int g) {
    Frag16 f;
    f.h[0] = *(const v8bf*)(row + 16 * g);
    f.h[1] = *(const v8bf*)(row + 16 * g + 8);
    return f.v;
}

// B-fragment (32x16 bf16, K x N) from a row-major [n][k] source:
// lane holds column n = lane&15, K contiguous 16g..16g+15.
__device__ inline v16bf load_frag_b(const bf16* row, int g) {
    return *(const v16bf*)(row + 16 * g);
}

// CDNA5 async global->LDS copy (ASYNCcnt-tracked, no VGPR data path).
// LDS byte offset = low 32 bits of the generic shared address (LDS aperture).
__device__ __forceinline__ void async_copy_b128(void* lds, const void* gaddr) {
    unsigned int lds_off = (unsigned int)(unsigned long long)lds;
    asm volatile("global_load_async_to_lds_b128 %0, %1, off"
                 :: "v"(lds_off), "v"(gaddr)
                 : "memory");
}
__device__ __forceinline__ void wait_async0() {
    asm volatile("s_wait_asynccnt 0" ::: "memory");
}

// ---------------------------------------------------------------------------
// Projection: Y = X(8192x1024) @ W^T(1024x1024), bf16 WMMA, fp32 accum.
// proj 0 -> Qh[B,H,S,HD], proj 1 -> Kh[B,H,S,HD], proj 2 -> Vt[B,H,HD,S].
// ---------------------------------------------------------------------------
__global__ __launch_bounds__(256) void proj_kernel(
    const float* __restrict__ Xq, const float* __restrict__ Xk, const float* __restrict__ Xv,
    const float* __restrict__ Wq, const float* __restrict__ Wk, const float* __restrict__ Wv,
    bf16* __restrict__ Qh, bf16* __restrict__ Kh, bf16* __restrict__ Vt)
{
    const int proj = blockIdx.z;
    const float* X = (proj == 0) ? Xq : (proj == 1) ? Xk : Xv;
    const float* W = (proj == 0) ? Wq : (proj == 1) ? Wk : Wv;

    __shared__ __align__(64) bf16 Xs[2][128][32];
    __shared__ __align__(64) bf16 Ws[2][128][32];

    const int tid  = threadIdx.x;
    const int lane = tid & 31;
    const int wv   = tid >> 5;   // wave 0..7
    const int wm   = wv >> 2;    // 0..1  (row group of 64)
    const int wn   = wv & 3;     // 0..3  (col group of 32)
    const int g    = lane >> 4;
    const int n16  = lane & 15;

    const int rBase = blockIdx.y * 128;
    const int cBase = blockIdx.x * 128;

    v8f acc[4][2];
    #pragma unroll
    for (int i = 0; i < 4; ++i)
        #pragma unroll
        for (int j = 0; j < 2; ++j) acc[i][j] = zero8();

    auto load_tile = [&](int kt, int buf) {
        const int k0 = kt * 32;
        #pragma unroll
        for (int i = 0; i < 4; ++i) {
            int vec = tid + 256 * i;   // 0..1023
            int row = vec >> 3;        // 0..127
            int cv  = vec & 7;         // 0..7 (float4 within 32-wide row)
            float4 xf = *(const float4*)(X + (size_t)(rBase + row) * Dm + k0 + cv * 4);
            float4 wf = *(const float4*)(W + (size_t)(cBase + row) * Dm + k0 + cv * 4);
            v4bf xb = { (bf16)xf.x, (bf16)xf.y, (bf16)xf.z, (bf16)xf.w };
            v4bf wb = { (bf16)wf.x, (bf16)wf.y, (bf16)wf.z, (bf16)wf.w };
            *(v4bf*)&Xs[buf][row][cv * 4] = xb;
            *(v4bf*)&Ws[buf][row][cv * 4] = wb;
        }
    };

    load_tile(0, 0);
    int buf = 0;
    for (int kt = 0; kt < 32; ++kt) {
        __syncthreads();
        if (kt + 1 < 32) load_tile(kt + 1, buf ^ 1);

        v16bf a[4];
        #pragma unroll
        for (int i = 0; i < 4; ++i)
            a[i] = load_frag_a(&Xs[buf][wm * 64 + i * 16 + n16][0], g);
        #pragma unroll
        for (int j = 0; j < 2; ++j) {
            v16bf b = load_frag_b(&Ws[buf][wn * 32 + j * 16 + n16][0], g);
            #pragma unroll
            for (int i = 0; i < 4; ++i)
                acc[i][j] = __builtin_amdgcn_wmma_f32_16x16x32_bf16(
                    false, a[i], false, b, (short)0, acc[i][j], false, false);
        }
        buf ^= 1;
    }

    // Epilogue: D[m][n] with m = v + 8g (VGPR v), n = lane&15.
    const int r0 = rBase + wm * 64 + 8 * g;       // + i*16 + v
    const int o0 = cBase + wn * 32 + n16;         // + j*16
    if (proj != 2) {
        bf16* __restrict__ dst = (proj == 0) ? Qh : Kh;
        #pragma unroll
        for (int i = 0; i < 4; ++i) {
            #pragma unroll
            for (int j = 0; j < 2; ++j) {
                #pragma unroll
                for (int v = 0; v < 8; ++v) {
                    int r = r0 + i * 16 + v;
                    int o = o0 + j * 16;
                    int b_ = r >> 10, s_ = r & 1023;
                    int h_ = o >> 6,  d_ = o & 63;
                    dst[(((size_t)b_ * Hh + h_) * Sseq + s_) * HDim + d_] =
                        (bf16)acc[i][j][v];
                }
            }
        }
    } else {
        #pragma unroll
        for (int i = 0; i < 4; ++i) {
            #pragma unroll
            for (int j = 0; j < 2; ++j) {
                #pragma unroll
                for (int v = 0; v < 8; ++v) {
                    int r = r0 + i * 16 + v;
                    int o = o0 + j * 16;
                    int b_ = r >> 10, s_ = r & 1023;
                    int h_ = o >> 6,  d_ = o & 63;
                    Vt[(((size_t)b_ * Hh + h_) * HDim + d_) * Sseq + s_] =
                        (bf16)acc[i][j][v];
                }
            }
        }
    }
}

// ---------------------------------------------------------------------------
// Flash attention: 128 queries per block (16 per wave), key blocks of 64.
// K/V tiles double-buffered in LDS via async global->LDS DMA (ASYNCcnt).
// ---------------------------------------------------------------------------
__global__ __launch_bounds__(256) void attn_kernel(
    const bf16* __restrict__ Qh, const bf16* __restrict__ Kh,
    const bf16* __restrict__ Vt, float* __restrict__ Out)
{
    __shared__ __align__(64) bf16 Ks[2][64][64];   // [key][hd]
    __shared__ __align__(64) bf16 Vs[2][64][64];   // [hd][key]  (from Vt)
    __shared__ __align__(64) bf16 Ps[8][16][64];   // per-wave P bounce [m][key]

    const int tid  = threadIdx.x;
    const int lane = tid & 31;
    const int wv   = tid >> 5;
    const int g    = lane >> 4;
    const int n16  = lane & 15;

    const int qb = blockIdx.x;           // query block (128)
    const int h  = blockIdx.y;
    const int b  = blockIdx.z;
    const size_t bh = (size_t)b * Hh + h;

    // Q fragments for this wave's 16 rows, pre-scaled by 1/sqrt(64)=0.125.
    const int q = qb * 128 + wv * 16 + n16;
    const bf16* qrow = Qh + (bh * Sseq + q) * HDim;
    v16bf qf[2];
    #pragma unroll
    for (int c = 0; c < 2; ++c) {
        Frag16 f;
        f.h[0] = *(const v8bf*)(qrow + 32 * c + 16 * g);
        f.h[1] = *(const v8bf*)(qrow + 32 * c + 16 * g + 8);
        #pragma unroll
        for (int e = 0; e < 16; ++e) f.v[e] = (bf16)((float)f.v[e] * 0.125f);
        qf[c] = f.v;
    }

    float m_run[8], l_run[8];
    v8f oacc[4];
    #pragma unroll
    for (int v = 0; v < 8; ++v) { m_run[v] = -__builtin_inff(); l_run[v] = 0.0f; }
    #pragma unroll
    for (int i = 0; i < 4; ++i) oacc[i] = zero8();

    const bf16* Kbase = Kh + bh * Sseq * HDim;
    const bf16* Vbase = Vt + bh * HDim * Sseq;

    // Async-stage one 64x64 K tile + 64x64 V^T tile (16B per lane-chunk).
    auto stage = [&](int kb2, int bb) {
        #pragma unroll
        for (int i = 0; i < 2; ++i) {
            int chunk = tid + 256 * i;   // 0..511
            int row = chunk >> 3;        // 0..63
            int c8  = chunk & 7;
            async_copy_b128(&Ks[bb][row][c8 * 8],
                            Kbase + (size_t)(kb2 * 64 + row) * HDim + c8 * 8);
            async_copy_b128(&Vs[bb][row][c8 * 8],
                            Vbase + (size_t)row * Sseq + kb2 * 64 + c8 * 8);
        }
    };

    stage(0, 0);
    int buf = 0;
    for (int kb = 0; kb < 16; ++kb) {
        wait_async0();       // this wave's DMA into `buf` has landed
        __syncthreads();     // ... and every other wave's too
        if (kb + 1 < 16) stage(kb + 1, buf ^ 1);   // prefetch next tile

        // scores = (Q*scale) @ K^T for 4 key tiles of 16
        v8f sc[4];
        #pragma unroll
        for (int j = 0; j < 4; ++j) {
            v8f s = zero8();
            #pragma unroll
            for (int c = 0; c < 2; ++c) {
                v16bf kf = *(const v16bf*)&Ks[buf][j * 16 + n16][32 * c + 16 * g];
                s = __builtin_amdgcn_wmma_f32_16x16x32_bf16(
                        false, qf[c], false, kf, (short)0, s, false, false);
            }
            sc[j] = s;
        }

        // online softmax: row m = v + 8g lives across the 16-lane half-wave
        float mnew[8], corr[8];
        #pragma unroll
        for (int v = 0; v < 8; ++v) {
            float t = fmaxf(fmaxf(sc[0][v], sc[1][v]), fmaxf(sc[2][v], sc[3][v]));
            #pragma unroll
            for (int msk = 1; msk < 16; msk <<= 1)
                t = fmaxf(t, __shfl_xor(t, msk, 32));
            mnew[v] = fmaxf(m_run[v], t);
        }
        #pragma unroll
        for (int v = 0; v < 8; ++v) {
            float r = 0.0f;
            #pragma unroll
            for (int j = 0; j < 4; ++j) {
                float p = __expf(sc[j][v] - mnew[v]);
                sc[j][v] = p;
                r += p;
            }
            #pragma unroll
            for (int msk = 1; msk < 16; msk <<= 1)
                r += __shfl_xor(r, msk, 32);
            corr[v]  = __expf(m_run[v] - mnew[v]);
            l_run[v] = l_run[v] * corr[v] + r;
            m_run[v] = mnew[v];
        }
        #pragma unroll
        for (int i = 0; i < 4; ++i)
            #pragma unroll
            for (int v = 0; v < 8; ++v) oacc[i][v] *= corr[v];

        // Bounce P through per-wave LDS: C-layout -> A-layout (intra-wave,
        // DScnt-ordered; no barrier needed).
        #pragma unroll
        for (int j = 0; j < 4; ++j)
            #pragma unroll
            for (int v = 0; v < 8; ++v)
                Ps[wv][v + 8 * g][j * 16 + n16] = (bf16)sc[j][v];

        v16bf pf[2];
        #pragma unroll
        for (int c = 0; c < 2; ++c) {
            Frag16 f;
            f.h[0] = *(const v8bf*)&Ps[wv][n16][32 * c + 16 * g];
            f.h[1] = *(const v8bf*)&Ps[wv][n16][32 * c + 16 * g + 8];
            pf[c] = f.v;
        }

        // oacc += P @ V   (B[k][n] = Vs[n][k], contiguous per lane)
        #pragma unroll
        for (int i = 0; i < 4; ++i) {
            #pragma unroll
            for (int c = 0; c < 2; ++c) {
                v16bf vf = *(const v16bf*)&Vs[buf][i * 16 + n16][32 * c + 16 * g];
                oacc[i] = __builtin_amdgcn_wmma_f32_16x16x32_bf16(
                              false, pf[c], false, vf, (short)0, oacc[i], false, false);
            }
        }
        buf ^= 1;
    }

    // Out[b][q][h*64+hd] = oacc / l
    #pragma unroll
    for (int i = 0; i < 4; ++i) {
        #pragma unroll
        for (int v = 0; v < 8; ++v) {
            int qq  = qb * 128 + wv * 16 + v + 8 * g;
            int col = h * HDim + i * 16 + n16;
            Out[((size_t)b * Sseq + qq) * Dm + col] = oacc[i][v] / l_run[v];
        }
    }
}

extern "C" void kernel_launch(void* const* d_in, const int* in_sizes, int n_in,
                              void* d_out, int out_size, void* d_ws, size_t ws_size,
                              hipStream_t stream) {
    const float* Qs = (const float*)d_in[0];
    const float* Kq = (const float*)d_in[1];
    const float* Vq = (const float*)d_in[2];
    const float* WQ = (const float*)d_in[3];
    const float* WK = (const float*)d_in[4];
    const float* WV = (const float*)d_in[5];
    float* Out = (float*)d_out;

    const size_t elems = (size_t)Bsz * Hh * Sseq * HDim;   // 8.4M bf16 each
    bf16* Qh = (bf16*)d_ws;
    bf16* Kh = Qh + elems;
    bf16* Vt = Kh + elems;

    dim3 pg(Dm / 128, (Bsz * Sseq) / 128, 3);
    proj_kernel<<<pg, 256, 0, stream>>>(Qs, Kq, Vq, WQ, WK, WV, Qh, Kh, Vt);

    dim3 ag(Sseq / 128, Hh, Bsz);
    attn_kernel<<<ag, 256, 0, stream>>>(Qh, Kh, Vt, Out);
}
